// ResDWBlock_1297080124173
// MI455X (gfx1250) — matrix operand order repaired and checked by reading, not compiled, not run on hardware
//
#include <hip/hip_runtime.h>
#include <hip/hip_bf16.h>

typedef __attribute__((ext_vector_type(16))) __bf16 bf16x16;
typedef __attribute__((ext_vector_type(8)))  float  f32x8;

#define PIX_PER_IMG 4096   // 64*64
#define C_DIM 512
#define HID   2048
#define NPIX  65536        // 16*4096
#define MT    64           // pixel tile per workgroup
#define ROW_AB 520         // 512 + 8 bf16 pad (row stride, bank-conflict-free frag loads)
#define ROW_C  2056        // 2048 + 8 bf16 pad

// d_ws layout
#define HID_BYTES   ((size_t)HID * NPIX * 2)          // 256 MB bf16 hidden
#define WC_ELEMS    ((size_t)C_DIM * C_DIM)           // 262144
#define W1_ELEMS    ((size_t)HID * C_DIM)             // 1048576
#define W2_ELEMS    ((size_t)C_DIM * HID)             // 1048576

__device__ __forceinline__ float gelu_f(float v) {
    return 0.5f * v * (1.0f + erff(v * 0.70710678118654752440f));
}

union PK8 { __bf16 h[8]; uint4 u; };
union PK4 { __bf16 h[4]; uint2 u; };

// A fragment: 16x32 bf16, lane<16: M=lane, K = kb+{0..7} and kb+{16..23};
// lane>=16: M=lane-16, K = kb+{8..15} and kb+{24..31}.  (ISA 7.12.2)
__device__ __forceinline__ bf16x16 load_a_frag(const __bf16* lds_a, int rowstride,
                                               int mbase, int kb, int lane) {
    const int m  = mbase + (lane & 15);
    const int k0 = kb + ((lane & 16) ? 8 : 0);
    const uint4* p = (const uint4*)(lds_a + m * rowstride + k0);
    union { uint4 u[2]; bf16x16 v; } r;
    r.u[0] = p[0];      // K = k0 .. k0+7
    r.u[1] = p[2];      // K = k0+16 .. k0+23 (+32 bytes)
    return r.v;
}

// B fragment: 32x16 bf16, lane<16: K = kb+{0..7},{16..23}, lane>=16: K = kb+{8..15},{24..31},
// column N = lane&15. Pre-converted bf16 weights, [O][K] row-major.
__device__ __forceinline__ bf16x16 load_b_frag(const __bf16* __restrict__ W, int ldk,
                                               int n, int kb, int lane) {
    const int k0 = kb + ((lane & 16) ? 8 : 0);
    const uint4* p = (const uint4*)(W + (size_t)n * ldk + k0);
    union { uint4 u[2]; bf16x16 v; } r;
    r.u[0] = p[0];
    r.u[1] = p[2];
    return r.v;
}

// fp32 -> bf16 weight conversion (runs once per launch; weights are tiny vs x)
__global__ void __launch_bounds__(256)
k_cvt_bf16(const float* __restrict__ src, __bf16* __restrict__ dst, int n4) {
    int i = blockIdx.x * 256 + threadIdx.x;
    if (i < n4) {
        float4 f = ((const float4*)src)[i];
        PK4 o;
        o.h[0] = (__bf16)f.x; o.h[1] = (__bf16)f.y;
        o.h[2] = (__bf16)f.z; o.h[3] = (__bf16)f.w;
        ((uint2*)dst)[i] = o.u;
    }
}

// Phase 0: per-pixel RMS over 512 channels, then write bf16(gelu?(x*rsqrt*w))
// into LDS A-tile [64][ROW_AB].  256 threads: pixel = t&63, quarter = t>>6.
template <bool DO_GELU>
__device__ void stage_norm_tile(const float* __restrict__ xb, const float* __restrict__ nw,
                                int hwbase, __bf16* lds_a, float* red, float* sscale) {
    const int t = threadIdx.x;
    const int m = t & 63;
    const int q = t >> 6;
    const int hw = hwbase + m;
    float s = 0.f;
#pragma unroll 1
    for (int c = q * 128; c < q * 128 + 128; ++c) {
        float v = xb[(size_t)c * PIX_PER_IMG + hw];
        s += v * v;
    }
    red[t] = s;
    __syncthreads();
    if (t < 64) {
        float tot = red[t] + red[t + 64] + red[t + 128] + red[t + 192];
        sscale[t] = rsqrtf(tot * (1.0f / 512.0f) + 1e-8f);
    }
    __syncthreads();
    const float scale = sscale[m];
#pragma unroll 1
    for (int c = q * 128; c < q * 128 + 128; c += 8) {
        PK8 pk;
#pragma unroll
        for (int j = 0; j < 8; ++j) {
            float v = xb[(size_t)(c + j) * PIX_PER_IMG + hw] * scale * nw[c + j];
            if (DO_GELU) v = gelu_f(v);
            pk.h[j] = (__bf16)v;
        }
        *(uint4*)&lds_a[m * ROW_AB + c] = pk.u;
    }
    __syncthreads();
}

// ---------------- Stage 1: x1 = x + conv1x1(gelu(rmsnorm(x))) ----------------
__global__ void __launch_bounds__(256)
k_stage1(const float* __restrict__ x, const float* __restrict__ n1w,
         const __bf16* __restrict__ Wc, const float* __restrict__ Bc,
         float* __restrict__ x1) {
    extern __shared__ char smem[];
    __bf16* lds_a  = (__bf16*)smem;
    float*  red    = (float*)(smem + (size_t)MT * ROW_AB * sizeof(__bf16));
    float*  sscale = red + 256;

    const int p0 = blockIdx.x * MT;
    const int b  = p0 >> 12;
    const int hwbase = p0 & 4095;
    const float* xb = x  + (size_t)b * C_DIM * PIX_PER_IMG;
    float*       ob = x1 + (size_t)b * C_DIM * PIX_PER_IMG;

    stage_norm_tile<true>(xb, n1w, hwbase, lds_a, red, sscale);

    const int wid = threadIdx.x >> 5;
    const int lane = threadIdx.x & 31;
    const f32x8 zero = {};
#pragma unroll 1
    for (int nb = wid; nb < C_DIM / 16; nb += 8) {      // 4 chunks/wave
        const int n = nb * 16 + (lane & 15);
        f32x8 acc[4] = {zero, zero, zero, zero};
#pragma unroll 1
        for (int ks = 0; ks < C_DIM; ks += 32) {
            bf16x16 bf = load_b_frag(Wc, C_DIM, n, ks, lane);
#pragma unroll
            for (int ms = 0; ms < 4; ++ms) {
                bf16x16 a = load_a_frag(lds_a, ROW_AB, ms * 16, ks, lane);
                acc[ms] = __builtin_amdgcn_wmma_f32_16x16x32_bf16(
                    false, a, false, bf, (short)0, acc[ms], false, false);
            }
        }
        const float bias = Bc[n];
        const int half = (lane & 16) ? 8 : 0;
#pragma unroll
        for (int ms = 0; ms < 4; ++ms) {
            size_t gi = (size_t)n * PIX_PER_IMG + hwbase + ms * 16 + half;
            float4 r0 = *(const float4*)&xb[gi];
            float4 r1 = *(const float4*)&xb[gi + 4];
            float4 o0, o1;
            o0.x = acc[ms][0] + bias + r0.x; o0.y = acc[ms][1] + bias + r0.y;
            o0.z = acc[ms][2] + bias + r0.z; o0.w = acc[ms][3] + bias + r0.w;
            o1.x = acc[ms][4] + bias + r1.x; o1.y = acc[ms][5] + bias + r1.y;
            o1.z = acc[ms][6] + bias + r1.z; o1.w = acc[ms][7] + bias + r1.w;
            *(float4*)&ob[gi]     = o0;
            *(float4*)&ob[gi + 4] = o1;
        }
    }
}

// ---------- Stage 2a: hidden = gelu(conv1x1(rmsnorm(x1), W1, b1)) (bf16) ----------
__global__ void __launch_bounds__(256)
k_stage2a(const float* __restrict__ x1, const float* __restrict__ n2w,
          const __bf16* __restrict__ W1, const float* __restrict__ B1,
          __bf16* __restrict__ hid) {
    extern __shared__ char smem[];
    __bf16* lds_a  = (__bf16*)smem;
    float*  red    = (float*)(smem + (size_t)MT * ROW_AB * sizeof(__bf16));
    float*  sscale = red + 256;

    const int p0 = blockIdx.x * MT;
    const int b  = p0 >> 12;
    const int hwbase = p0 & 4095;
    const float* xb = x1 + (size_t)b * C_DIM * PIX_PER_IMG;

    stage_norm_tile<false>(xb, n2w, hwbase, lds_a, red, sscale);

    const int wid = threadIdx.x >> 5;
    const int lane = threadIdx.x & 31;
    const f32x8 zero = {};
#pragma unroll 1
    for (int nb = wid; nb < HID / 16; nb += 8) {        // 16 chunks/wave
        const int n = nb * 16 + (lane & 15);
        f32x8 acc[4] = {zero, zero, zero, zero};
#pragma unroll 1
        for (int ks = 0; ks < C_DIM; ks += 32) {
            bf16x16 bf = load_b_frag(W1, C_DIM, n, ks, lane);
#pragma unroll
            for (int ms = 0; ms < 4; ++ms) {
                bf16x16 a = load_a_frag(lds_a, ROW_AB, ms * 16, ks, lane);
                acc[ms] = __builtin_amdgcn_wmma_f32_16x16x32_bf16(
                    false, a, false, bf, (short)0, acc[ms], false, false);
            }
        }
        const float bias = B1[n];
        const int half = (lane & 16) ? 8 : 0;
#pragma unroll
        for (int ms = 0; ms < 4; ++ms) {
            PK8 pk;
#pragma unroll
            for (int r = 0; r < 8; ++r)
                pk.h[r] = (__bf16)gelu_f(acc[ms][r] + bias);
            // hidden stored channel-major: [HID][NPIX]
            size_t hi = (size_t)n * NPIX + p0 + ms * 16 + half;
            *(uint4*)&hid[hi] = pk.u;
        }
    }
}

// ---------- Stage 2b: out = x1 + conv1x1(hidden, W2, b2); in-place on d_out ----------
__global__ void __launch_bounds__(256)
k_stage2b(const __bf16* __restrict__ hid, const __bf16* __restrict__ W2,
          const float* __restrict__ B2, float* __restrict__ xio) {
    extern __shared__ char smem[];
    __bf16* lds_a = (__bf16*)smem;

    const int p0 = blockIdx.x * MT;
    const int b  = p0 >> 12;
    const int hwbase = p0 & 4095;
    float* ob = xio + (size_t)b * C_DIM * PIX_PER_IMG;

    // Stage full 64x2048 bf16 hidden tile into LDS (257 KB of 320 KB WGP LDS)
    {
        const int t = threadIdx.x, m = t & 63, q = t >> 6;
        const size_t p = (size_t)p0 + m;
#pragma unroll 1
        for (int c = q * 512; c < q * 512 + 512; c += 8) {
            PK8 pk;
#pragma unroll
            for (int j = 0; j < 8; ++j)
                pk.h[j] = hid[(size_t)(c + j) * NPIX + p];
            *(uint4*)&lds_a[m * ROW_C + c] = pk.u;
        }
    }
    __syncthreads();

    const int wid = threadIdx.x >> 5;
    const int lane = threadIdx.x & 31;
    const f32x8 zero = {};
#pragma unroll 1
    for (int nb = wid; nb < C_DIM / 16; nb += 8) {      // 4 chunks/wave
        const int n = nb * 16 + (lane & 15);
        f32x8 acc[4] = {zero, zero, zero, zero};
#pragma unroll 1
        for (int ks = 0; ks < HID; ks += 32) {          // K = 2048
            bf16x16 bf = load_b_frag(W2, HID, n, ks, lane);
#pragma unroll
            for (int ms = 0; ms < 4; ++ms) {
                bf16x16 a = load_a_frag(lds_a, ROW_C, ms * 16, ks, lane);
                acc[ms] = __builtin_amdgcn_wmma_f32_16x16x32_bf16(
                    false, a, false, bf, (short)0, acc[ms], false, false);
            }
        }
        const float bias = B2[n];
        const int half = (lane & 16) ? 8 : 0;
#pragma unroll
        for (int ms = 0; ms < 4; ++ms) {
            size_t gi = (size_t)n * PIX_PER_IMG + hwbase + ms * 16 + half;
            float4 r0 = *(const float4*)&ob[gi];        // x1 residual (read-then-write, same thread)
            float4 r1 = *(const float4*)&ob[gi + 4];
            float4 o0, o1;
            o0.x = acc[ms][0] + bias + r0.x; o0.y = acc[ms][1] + bias + r0.y;
            o0.z = acc[ms][2] + bias + r0.z; o0.w = acc[ms][3] + bias + r0.w;
            o1.x = acc[ms][4] + bias + r1.x; o1.y = acc[ms][5] + bias + r1.y;
            o1.z = acc[ms][6] + bias + r1.z; o1.w = acc[ms][7] + bias + r1.w;
            *(float4*)&ob[gi]     = o0;
            *(float4*)&ob[gi + 4] = o1;
        }
    }
}

extern "C" void kernel_launch(void* const* d_in, const int* in_sizes, int n_in,
                              void* d_out, int out_size, void* d_ws, size_t ws_size,
                              hipStream_t stream) {
    const float* x   = (const float*)d_in[0];
    const float* n1w = (const float*)d_in[1];
    const float* Wc  = (const float*)d_in[2];
    const float* Bc  = (const float*)d_in[3];
    const float* n2w = (const float*)d_in[4];
    const float* W1  = (const float*)d_in[5];
    const float* B1  = (const float*)d_in[6];
    const float* W2  = (const float*)d_in[7];
    const float* B2  = (const float*)d_in[8];
    float*  xout = (float*)d_out;                    // x1 after stage1, final out after 2b
    __bf16* hid  = (__bf16*)d_ws;                    // [2048][65536] bf16 = 256 MB
    __bf16* wcb  = (__bf16*)((char*)d_ws + HID_BYTES);
    __bf16* w1b  = wcb + WC_ELEMS;
    __bf16* w2b  = w1b + W1_ELEMS;

    dim3 blk(256);
    // weight pre-conversion to bf16 (cheap; weights are L2-resident afterwards)
    k_cvt_bf16<<<dim3((WC_ELEMS / 4 + 255) / 256), blk, 0, stream>>>(Wc, wcb, WC_ELEMS / 4);
    k_cvt_bf16<<<dim3((W1_ELEMS / 4 + 255) / 256), blk, 0, stream>>>(W1, w1b, W1_ELEMS / 4);
    k_cvt_bf16<<<dim3((W2_ELEMS / 4 + 255) / 256), blk, 0, stream>>>(W2, w2b, W2_ELEMS / 4);

    dim3 grid(NPIX / MT);                            // 1024 workgroups
    size_t smemAB = (size_t)MT * ROW_AB * sizeof(__bf16) + 256 * 4 + 64 * 4;  // ~67.8 KB
    size_t smemC  = (size_t)MT * ROW_C  * sizeof(__bf16);                     // ~257 KB

    k_stage1 <<<grid, blk, smemAB, stream>>>(x, n1w, wcb, Bc, xout);
    k_stage2a<<<grid, blk, smemAB, stream>>>(xout, n2w, w1b, B1, hid);
    k_stage2b<<<grid, blk, smemC,  stream>>>(hid, w2b, B2, xout);
}
